// Attention_59966333387168
// MI455X (gfx1250) — matrix operand order repaired and checked by reading, not compiled, not run on hardware
//
#include <hip/hip_runtime.h>
#include <hip/hip_bf16.h>

// bf16 WMMA fragments (wave32, 16x16x32): A = 16 bf16/lane, C/D = 8 f32/lane.
typedef __attribute__((ext_vector_type(16))) __bf16 v16bf;
typedef __attribute__((ext_vector_type(8)))  float  v8f;

union Frag16 {
  v16bf v;
  uint4 q[2];
  unsigned int u[8];
  unsigned short h[16];
};

__device__ __forceinline__ unsigned short f2bf(float f) {
  unsigned int b = __float_as_uint(f);
  b += 0x7FFFu + ((b >> 16) & 1u);   // round-to-nearest-even
  return (unsigned short)(b >> 16);
}

__device__ __forceinline__ v8f wmma_bf16(const Frag16& a, const Frag16& b, v8f c) {
  return __builtin_amdgcn_wmma_f32_16x16x32_bf16(false, a.v, false, b.v,
                                                 (short)0, c, false, false);
}

// ---- elementwise f32 -> bf16 (4 elems/thread) -------------------------------
__global__ void cvt_bf16_kernel(const float* __restrict__ in,
                                unsigned short* __restrict__ out, int n4) {
  int i = blockIdx.x * blockDim.x + threadIdx.x;
  int stride = gridDim.x * blockDim.x;
  for (; i < n4; i += stride) {
    float4 f = ((const float4*)in)[i];
    ushort4 o;
    o.x = f2bf(f.x); o.y = f2bf(f.y); o.z = f2bf(f.z); o.w = f2bf(f.w);
    ((ushort4*)out)[i] = o;
  }
}

// ---- W[K][N] f32 -> Wt[N][K] bf16 (contiguous-K for B fragments) ------------
__global__ void transpose_bf16_kernel(const float* __restrict__ in,
                                      unsigned short* __restrict__ out,
                                      int K, int N) {
  int i = blockIdx.x * blockDim.x + threadIdx.x;
  int stride = gridDim.x * blockDim.x;
  int total = K * N;
  for (; i < total; i += stride) {
    int n = i / K, kk = i - n * K;
    out[i] = f2bf(in[kk * N + n]);
  }
}

// ---- projection GEMM: Y[m][n] = sum_k xb[m][k]*Wt[n][k] + bias[n] -----------
// 32x64 output per wave (2 M-subtiles x 4 N-subtiles) -> 1.5 vmem/WMMA.
// Output address: m*sm + n*sn + (m>>11)*sb  (lets us write V transposed per batch).
__global__ __launch_bounds__(256) void proj_gemm_kernel(
    const unsigned short* __restrict__ xb,   // [16384][1024] bf16
    const unsigned short* __restrict__ wt,   // [N][1024] bf16
    const float* __restrict__ bias,          // [N]
    unsigned short* __restrict__ out,
    int N, int sm, int sn, int sb) {
  const int K = 1024;
  int lane = threadIdx.x & 31;
  int hlf  = lane >> 4;
  int l15  = lane & 15;
  int wave = blockIdx.x * (blockDim.x >> 5) + (threadIdx.x >> 5);
  int tiles_n = N >> 6;                        // 64-wide N tiles
  int tm = wave / tiles_n, tn = wave - tm * tiles_n;
  int m0 = tm << 5, n0 = tn << 6;

  const unsigned short* arow0 = xb + (size_t)(m0 + l15) * K;
  const unsigned short* arow1 = arow0 + (size_t)16 * K;
  const unsigned short* brow  = wt + (size_t)(n0 + l15) * K + hlf * 16;

  v8f acc[2][4];
#pragma unroll
  for (int i = 0; i < 2; ++i)
#pragma unroll
    for (int j = 0; j < 4; ++j) acc[i][j] = (v8f){};

  for (int kk0 = 0; kk0 < K; kk0 += 32) {
    Frag16 a0, a1;
#pragma unroll
    for (int i = 0; i < 8; ++i) {
      int kp = kk0 + 2 * i + hlf * 16;  // A K-contiguous
      a0.u[i] = *(const unsigned int*)(arow0 + kp);
      a1.u[i] = *(const unsigned int*)(arow1 + kp);
    }
#pragma unroll
    for (int j = 0; j < 4; ++j) {
      Frag16 b;
      const unsigned short* bp = brow + (size_t)(j * 16) * K + kk0;
      b.q[0] = *(const uint4*)(bp);
      b.q[1] = *(const uint4*)(bp + 8);
      acc[0][j] = wmma_bf16(a0, b, acc[0][j]);
      acc[1][j] = wmma_bf16(a1, b, acc[1][j]);
    }
  }
#pragma unroll
  for (int j = 0; j < 4; ++j) {
    int n = n0 + j * 16 + l15;
    float bval = bias[n];
#pragma unroll
    for (int i = 0; i < 2; ++i)
#pragma unroll
      for (int r = 0; r < 8; ++r) {
        int m = m0 + i * 16 + r + hlf * 8;   // C layout: VGPR r -> row r / r+8
        out[(size_t)m * sm + (size_t)n * sn + (size_t)(m >> 11) * sb] =
            f2bf(acc[i][j][r] + bval);
      }
  }
}

// ---- fused flash attention + tanh -------------------------------------------
// Block = 16 waves (512 threads), 32 query rows (two 16-row tiles).
// Wave w: row-tile = w>>3, key-chunk/V-block = w&7.
// Keys processed in 256-key superblocks: wave w computes scores ONLY for keys
// [T0+32*(w&7), +32) of its row tile (no redundant score WMMAs); row max/sum
// combined via LDS; P (2 x 16x256 bf16) shared in LDS; each wave then runs
// 64 P@V WMMAs for its own 128 V-columns. Two row-tiles per block halve the
// per-block V/K re-read traffic from L2 (dup loads hit the shared WGP$).
__global__ __launch_bounds__(512) void flash_attn_kernel(
    const unsigned short* __restrict__ q,    // [16384][128] bf16
    const unsigned short* __restrict__ k,    // [16384][128] bf16
    const unsigned short* __restrict__ vt,   // [8][1024][2048] bf16 (V transposed)
    float* __restrict__ out) {               // [16384][1024] f32
  __shared__ unsigned short pbuf[2][16][256]; // shared P tiles, 16 KB
  __shared__ float redmax[16][16];            // per-wave partial row max
  __shared__ float redsum[16][16];            // per-wave partial row sum
  const int S = 2048, D = 128, V = 1024;
  int lane = threadIdx.x & 31;
  int hlf  = lane >> 4;
  int l15  = lane & 15;
  int w16  = threadIdx.x >> 5;        // wave id 0..15
  int tile = w16 >> 3;                // row-tile 0/1
  int wv   = w16 & 7;                 // key-chunk id = V-block id
  int b    = blockIdx.x >> 6;
  int mt   = blockIdx.x & 63;
  int m0   = b * S + mt * 32 + tile * 16;   // this wave's query-row base

  // Q fragments for own row tile: 16x128 as four 16x32 A-slices, loaded once.
  Frag16 aq[4];
  {
    const unsigned short* qrow = q + (size_t)(m0 + l15) * D;
#pragma unroll
    for (int s4 = 0; s4 < 4; ++s4)
#pragma unroll
      for (int i = 0; i < 8; ++i) {
        int kp = s4 * 32 + 2 * i + hlf * 16;
        aq[s4].u[i] = *(const unsigned int*)(qrow + kp);
      }
  }

  float mrow[8], lrow[8];
  v8f oacc[8];
#pragma unroll
  for (int r = 0; r < 8; ++r) { mrow[r] = -3.0e38f; lrow[r] = 0.0f; }
#pragma unroll
  for (int s = 0; s < 8; ++s) oacc[s] = (v8f){};

  const unsigned short* kbase = k  + (size_t)b * S * D;
  const unsigned short* vbase = vt + (size_t)b * V * S;

#pragma unroll 1
  for (int T0 = 0; T0 < S; T0 += 256) {
    int tw = T0 + wv * 32;            // this wave's 32-key score chunk
    // ---- scores for own chunk x own row tile: two 16x16 tiles ----
    v8f s0 = {}, s1 = {};
#pragma unroll
    for (int s4 = 0; s4 < 4; ++s4) {
      const unsigned short* kp0 = kbase + (size_t)(tw + l15) * D + s4 * 32 + hlf * 16;
      Frag16 bk0, bk1;
      bk0.q[0] = *(const uint4*)(kp0);
      bk0.q[1] = *(const uint4*)(kp0 + 8);
      bk1.q[0] = *(const uint4*)(kp0 + 16 * D);
      bk1.q[1] = *(const uint4*)(kp0 + 16 * D + 8);
      s0 = wmma_bf16(aq[s4], bk0, s0);
      s1 = wmma_bf16(aq[s4], bk1, s1);
    }
    // ---- local row max -> LDS ----
#pragma unroll
    for (int r = 0; r < 8; ++r) {
      float mx = fmaxf(s0[r], s1[r]);
      mx = fmaxf(mx, __shfl_xor(mx, 1, 32));
      mx = fmaxf(mx, __shfl_xor(mx, 2, 32));
      mx = fmaxf(mx, __shfl_xor(mx, 4, 32));
      mx = fmaxf(mx, __shfl_xor(mx, 8, 32));
      if (l15 == 0) redmax[w16][r + hlf * 8] = mx;
    }
    __syncthreads();
    // ---- block row max (over the 8 waves of own tile), exp, P -> LDS ----
    float sc[8];
    int wbase = tile * 8;
#pragma unroll
    for (int r = 0; r < 8; ++r) {
      int row = r + hlf * 8;
      float bmx = redmax[wbase][row];
#pragma unroll
      for (int w = 1; w < 8; ++w) bmx = fmaxf(bmx, redmax[wbase + w][row]);
      float mnew = fmaxf(mrow[r], bmx);
      sc[r] = __expf(mrow[r] - mnew);
      mrow[r] = mnew;
      float p0 = __expf(s0[r] - mnew);
      float p1 = __expf(s1[r] - mnew);
      float rs = p0 + p1;
      rs += __shfl_xor(rs, 1, 32);
      rs += __shfl_xor(rs, 2, 32);
      rs += __shfl_xor(rs, 4, 32);
      rs += __shfl_xor(rs, 8, 32);
      if (l15 == 0) redsum[w16][row] = rs;
      pbuf[tile][row][wv * 32 + l15]      = f2bf(p0);
      pbuf[tile][row][wv * 32 + 16 + l15] = f2bf(p1);
    }
    __syncthreads();
    // ---- combine sums, rescale O ----
#pragma unroll
    for (int r = 0; r < 8; ++r) {
      int row = r + hlf * 8;
      float rs = redsum[wbase][row];
#pragma unroll
      for (int w = 1; w < 8; ++w) rs += redsum[wbase + w][row];
      lrow[r] = lrow[r] * sc[r] + rs;
    }
#pragma unroll
    for (int s = 0; s < 8; ++s)
#pragma unroll
      for (int r = 0; r < 8; ++r) oacc[s][r] *= sc[r];

    // ---- O += P (16x256) @ V (256x128 slice): 8 key-chunks x 8 col-groups ----
#pragma unroll 1
    for (int c = 0; c < 8; ++c) {
      Frag16 pa;
#pragma unroll
      for (int i = 0; i < 8; ++i) {
        int kp = 2 * i + hlf * 16;
        pa.u[i] = *(const unsigned int*)&pbuf[tile][l15][c * 32 + kp];
      }
#pragma unroll
      for (int s = 0; s < 8; ++s) {
        int vcol = wv * 128 + s * 16 + l15;
        const unsigned short* vp = vbase + (size_t)vcol * S + T0 + c * 32 + hlf * 16;
        Frag16 bv;
        bv.q[0] = *(const uint4*)(vp);
        bv.q[1] = *(const uint4*)(vp + 8);
        oacc[s] = wmma_bf16(pa, bv, oacc[s]);
      }
    }
    __syncthreads();   // protect pbuf/red arrays before next superblock
  }

  float inv[8];
#pragma unroll
  for (int r = 0; r < 8; ++r) inv[r] = 1.0f / lrow[r];
#pragma unroll
  for (int s = 0; s < 8; ++s)
#pragma unroll
    for (int r = 0; r < 8; ++r) {
      int m = m0 + r + hlf * 8;
      int vcol = wv * 128 + s * 16 + l15;
      out[(size_t)m * V + vcol] = tanhf(oacc[s][r] * inv[r]);
    }
}

// -----------------------------------------------------------------------------
extern "C" void kernel_launch(void* const* d_in, const int* in_sizes, int n_in,
                              void* d_out, int out_size, void* d_ws, size_t ws_size,
                              hipStream_t stream) {
  const float* x  = (const float*)d_in[0];
  const float* Wq = (const float*)d_in[1];
  const float* bq = (const float*)d_in[2];
  const float* Wk = (const float*)d_in[3];
  const float* bk = (const float*)d_in[4];
  const float* Wv = (const float*)d_in[5];
  const float* bv = (const float*)d_in[6];
  float* out = (float*)d_out;

  const size_t M = 16384, K = 1024, QD = 128, VD = 1024;
  // workspace layout (~78.1 MB total)
  unsigned char* ws = (unsigned char*)d_ws;
  unsigned short* xb  = (unsigned short*)ws;  ws += M  * K * 2;   // x  bf16 [M][K]
  unsigned short* wqt = (unsigned short*)ws;  ws += QD * K * 2;   // Wq^T bf16 [128][1024]
  unsigned short* wkt = (unsigned short*)ws;  ws += QD * K * 2;   // Wk^T bf16
  unsigned short* wvt = (unsigned short*)ws;  ws += VD * K * 2;   // Wv^T bf16 [1024][1024]
  unsigned short* qb  = (unsigned short*)ws;  ws += M * QD * 2;   // q bf16 [M][128]
  unsigned short* kb  = (unsigned short*)ws;  ws += M * QD * 2;   // k bf16 [M][128]
  unsigned short* vtb = (unsigned short*)ws;  ws += M * VD * 2;   // V^T bf16 [8][1024][2048]

  cvt_bf16_kernel<<<2048, 256, 0, stream>>>(x, xb, (int)(M * K / 4));
  transpose_bf16_kernel<<<512,  256, 0, stream>>>(Wq, wqt, (int)K, (int)QD);
  transpose_bf16_kernel<<<512,  256, 0, stream>>>(Wk, wkt, (int)K, (int)QD);
  transpose_bf16_kernel<<<2048, 256, 0, stream>>>(Wv, wvt, (int)K, (int)VD);

  // q/k: (16384/32)*(128/64) = 1024 tiles -> 128 blocks of 8 waves
  proj_gemm_kernel<<<128, 256, 0, stream>>>(xb, wqt, bq, qb, (int)QD, 128, 1, 0);
  proj_gemm_kernel<<<128, 256, 0, stream>>>(xb, wkt, bk, kb, (int)QD, 128, 1, 0);
  // v: (16384/32)*(1024/64) = 8192 tiles -> 1024 blocks; output transposed per batch:
  // addr = m + n*2048 + (m>>11)*(2048*1024 - 2048)
  proj_gemm_kernel<<<1024, 256, 0, stream>>>(xb, wvt, bv, vtb, (int)VD,
                                             1, 2048, 2048 * 1024 - 2048);

  // 8 batches * 64 row-tile-pairs = 512 blocks; 16 waves (512 threads) each
  flash_attn_kernel<<<512, 512, 0, stream>>>(qb, kb, vtb, out);
}
